// ProteinEncoder_91259465105740
// MI455X (gfx1250) — compile-verified, matrix operand order checked by reading, not run on hardware
//
#include <hip/hip_runtime.h>
#include <hip/hip_bf16.h>
#include <float.h>

#define B_      2
#define L_      2048
#define K_      32
#define NRBF_   16
#define NODE_F_ 128
#define EDGE_F_ 128
#define MAX_REL_ 32
#define NODE_IN_ 35
#define EDGE_IN_ 468
#define EPAD    480   // EDGE_IN padded to 15*32 for WMMA K-steps
#define ARS     488   // sA row stride (elements): 16B aligned, 16 distinct banks over lanes

typedef __attribute__((ext_vector_type(16))) __bf16 v16bf;
typedef __attribute__((ext_vector_type(8)))  float  v8f;
typedef __attribute__((ext_vector_type(4)))  float  f32x4;

union ABfrag { v16bf v; f32x4 f[2]; };

__device__ inline unsigned short f2bf(float x) {
  union { float f; unsigned u; } v; v.f = x;
  unsigned r = v.u + 0x7FFFu + ((v.u >> 16) & 1u);  // RNE
  return (unsigned short)(r >> 16);
}

__device__ inline float dot3p(const float* a, const float* b) {
  return a[0]*b[0] + a[1]*b[1] + a[2]*b[2];
}
__device__ inline void cross3p(const float* a, const float* b, float* o) {
  o[0] = a[1]*b[2] - a[2]*b[1];
  o[1] = a[2]*b[0] - a[0]*b[2];
  o[2] = a[0]*b[1] - a[1]*b[0];
}

// matches reference _dihedral incl. nan_to_num semantics
__device__ float dihedral_f(const float* p0, const float* p1,
                            const float* p2, const float* p3) {
  float u0[3], u1[3], u2[3];
  for (int c = 0; c < 3; ++c) {
    u0[c] = p2[c] - p1[c];
    u1[c] = p0[c] - p1[c];
    u2[c] = p3[c] - p2[c];
  }
  float n1[3], n2[3], c12[3];
  cross3p(u0, u1, n1);
  cross3p(u0, u2, n2);
  cross3p(u1, u2, c12);
  float l1 = sqrtf(dot3p(n1, n1));
  float l2 = sqrtf(dot3p(n2, n2));
  float i1 = (l1 > 0.f) ? 1.f / l1 : 0.f;   // nan_to_num(normalize)
  float i2 = (l2 > 0.f) ? 1.f / l2 : 0.f;
  float d  = dot3p(n1, n2) * i1 * i2;
  float sg = dot3p(c12, u0);
  float sgn = (sg > 0.f) ? 1.f : ((sg < 0.f) ? -1.f : 0.f);
  float r = sgn * acosf(d);
  return (r == r) ? r : 0.f;                 // nan_to_num
}

// ---------------- W_edge -> bf16, transposed to [N=128][Kpad=480] ----------
__global__ void wconv_kernel(const float* __restrict__ We,
                             unsigned short* __restrict__ wT) {
  int idx = blockIdx.x * blockDim.x + threadIdx.x;
  if (idx >= EDGE_F_ * EPAD) return;
  int n = idx / EPAD, k = idx % EPAD;
  float v = (k < EDGE_IN_) ? We[k * EDGE_F_ + n] : 0.f;
  wT[n * EPAD + k] = f2bf(v);
}

// ---------------- top-K smallest D_adj per row (matches lax.top_k ties) ----
__global__ __launch_bounds__(256) void topk_kernel(
    const float* __restrict__ X, const float* __restrict__ mask,
    int* __restrict__ eidx, float* __restrict__ eidxf) {
  __shared__ float sD[L_];
  __shared__ float rV[256];
  __shared__ int   rI[256];
  int bi = blockIdx.x;
  int b = bi / L_;
  int t = threadIdx.x;
  float cax = X[(bi*4 + 1)*3 + 0];
  float cay = X[(bi*4 + 1)*3 + 1];
  float caz = X[(bi*4 + 1)*3 + 2];
  float mi = mask[bi];
  float lmax = -FLT_MAX;
  for (int j = t; j < L_; j += 256) {
    int bj = b * L_ + j;
    float dx = cax - X[(bj*4 + 1)*3 + 0];
    float dy = cay - X[(bj*4 + 1)*3 + 1];
    float dz = caz - X[(bj*4 + 1)*3 + 2];
    float m2 = mi * mask[bj];
    float D = m2 * sqrtf(dx*dx + dy*dy + dz*dz + 1e-6f);
    sD[j] = D;
    lmax = fmaxf(lmax, D);
  }
  rV[t] = lmax;
  __syncthreads();
  for (int s = 128; s > 0; s >>= 1) {
    if (t < s) rV[t] = fmaxf(rV[t], rV[t + s]);
    __syncthreads();
  }
  float rowmax = rV[0];
  __syncthreads();
  for (int j = t; j < L_; j += 256) {
    float m2 = mi * mask[b * L_ + j];
    sD[j] += 2.0f * (1.0f - m2) * rowmax;
  }
  __syncthreads();
  for (int sel = 0; sel < K_; ++sel) {
    float bv = FLT_MAX; int bj = L_;
    for (int j = t; j < L_; j += 256) {
      float v = sD[j];
      if (v < bv || (v == bv && j < bj)) { bv = v; bj = j; }
    }
    rV[t] = bv; rI[t] = bj;
    __syncthreads();
    for (int s = 128; s > 0; s >>= 1) {
      if (t < s) {
        float v2 = rV[t + s]; int j2 = rI[t + s];
        if (v2 < rV[t] || (v2 == rV[t] && j2 < rI[t])) { rV[t] = v2; rI[t] = j2; }
      }
      __syncthreads();
    }
    if (t == 0) {
      int win = rI[0];
      sD[win] = FLT_MAX;
      eidx[bi * K_ + sel]  = win;
      eidxf[bi * K_ + sel] = (float)win;
    }
    __syncthreads();
  }
}

// ---------------- node features + tiny GEMM + LayerNorm --------------------
__global__ __launch_bounds__(128) void node_kernel(
    const int* __restrict__ S, const float* __restrict__ BB,
    const float* __restrict__ SC, const float* __restrict__ Wn,
    const float* __restrict__ bn, const float* __restrict__ gn,
    const float* __restrict__ betan, float* __restrict__ hV) {
  __shared__ float sV[NODE_IN_];
  __shared__ float sS1[128], sS2[128];
  int bi = blockIdx.x, t = threadIdx.x;
  if (t < 21)      sV[t] = (S[bi] == t) ? 1.f : 0.f;
  else if (t < 27) sV[t] = BB[bi*6 + (t - 21)];
  else if (t < 35) sV[t] = SC[bi*8 + (t - 27)];
  __syncthreads();
  float acc = bn[t];
  for (int k = 0; k < NODE_IN_; ++k) acc += sV[k] * Wn[k * NODE_F_ + t];
  sS1[t] = acc; sS2[t] = acc * acc;
  __syncthreads();
  for (int s = 64; s > 0; s >>= 1) {
    if (t < s) { sS1[t] += sS1[t + s]; sS2[t] += sS2[t + s]; }
    __syncthreads();
  }
  float mean = sS1[0] * (1.f / 128.f);
  float var  = sS2[0] * (1.f / 128.f) - mean * mean;
  float inv  = rsqrtf(var + 1e-5f);
  hV[(size_t)bi * NODE_F_ + t] = (acc - mean) * inv * gn[t] + betan[t];
}

// ---------------- X passthrough -------------------------------------------
__global__ void copyx_kernel(const float* __restrict__ X,
                             float* __restrict__ outX, int n) {
  int idx = blockIdx.x * blockDim.x + threadIdx.x;
  if (idx < n) outX[idx] = X[idx];
}

// ---------------- fused edge features + WMMA GEMM + bias + LayerNorm ------
__global__ __launch_bounds__(512) void edge_kernel(
    const float* __restrict__ X, const int* __restrict__ chain,
    const int* __restrict__ eidx, const unsigned short* __restrict__ wT,
    const float* __restrict__ be, const float* __restrict__ ge,
    const float* __restrict__ betae, float* __restrict__ hE) {
  __shared__ __align__(16) unsigned short sA[K_ * ARS];   // 32 x 480 bf16 (stride 488)
  __shared__ float sC[K_ * EDGE_F_];                       // 32 x 128 f32
  __shared__ float sXi[5][3];
  __shared__ float sXnb[K_][5][3];
  __shared__ float sDn[K_][25];
  __shared__ int   sIdx[K_];
  __shared__ int   sCh[K_];

  int bi = blockIdx.x;
  int b = bi / L_, i = bi % L_;
  int tid = threadIdx.x;

  // Build X5 (N,Ca,C,O,Cb) for node i and its 32 neighbors
  if (tid < K_) {
    int j = eidx[bi * K_ + tid];
    sIdx[tid] = j;
    sCh[tid]  = chain[b * L_ + j];
    int bj = b * L_ + j;
    float nx[4][3];
    for (int a = 0; a < 4; ++a)
      for (int c = 0; c < 3; ++c)
        nx[a][c] = X[(bj*4 + a)*3 + c];
    float bv[3], cv[3], cr[3];
    for (int c = 0; c < 3; ++c) { bv[c] = nx[1][c] - nx[0][c]; cv[c] = nx[2][c] - nx[1][c]; }
    cross3p(bv, cv, cr);
    for (int a = 0; a < 4; ++a)
      for (int c = 0; c < 3; ++c) sXnb[tid][a][c] = nx[a][c];
    for (int c = 0; c < 3; ++c)
      sXnb[tid][4][c] = -0.58273431f*cr[c] + 0.56802827f*bv[c] - 0.54067466f*cv[c] + nx[1][c];
  } else if (tid == K_) {
    float nx[4][3];
    for (int a = 0; a < 4; ++a)
      for (int c = 0; c < 3; ++c)
        nx[a][c] = X[(bi*4 + a)*3 + c];
    float bv[3], cv[3], cr[3];
    for (int c = 0; c < 3; ++c) { bv[c] = nx[1][c] - nx[0][c]; cv[c] = nx[2][c] - nx[1][c]; }
    cross3p(bv, cv, cr);
    for (int a = 0; a < 4; ++a)
      for (int c = 0; c < 3; ++c) sXi[a][c] = nx[a][c];
    for (int c = 0; c < 3; ++c)
      sXi[4][c] = -0.58273431f*cr[c] + 0.56802827f*bv[c] - 0.54067466f*cv[c] + nx[1][c];
  }
  __syncthreads();

  // 25 atom-pair distances per edge
  for (int idx = tid; idx < K_ * 25; idx += 512) {
    int e = idx / 25, pq = idx % 25, p = pq / 5, q = pq % 5;
    float dx = sXi[p][0] - sXnb[e][q][0];
    float dy = sXi[p][1] - sXnb[e][q][1];
    float dz = sXi[p][2] - sXnb[e][q][2];
    sDn[e][pq] = sqrtf(dx*dx + dy*dy + dz*dz + 1e-6f);
  }
  __syncthreads();

  int ch_i = chain[bi];
  // 468 edge features (+ zero pad to 480), stored bf16 for WMMA
  for (int idx = tid; idx < K_ * EPAD; idx += 512) {
    int e = idx / EPAD, f = idx % EPAD;
    float val = 0.f;
    if (f < 65) {
      int off = sIdx[e] - i + MAX_REL_;
      off = off < 0 ? 0 : (off > 64 ? 64 : off);
      val = (f == off) ? 1.f : 0.f;
    } else if (f < 465) {
      int g = f - 65; int pq = g >> 4; int r = g & 15;
      float mu = (20.f / 15.f) * (float)r;     // linspace(0,20,16)
      float z = (sDn[e][pq] - mu) * 0.8f;      // sigma = 1.25
      val = __expf(-z * z);
    } else if (f == 465) {
      val = (sCh[e] == ch_i ? 1.f : 0.f) + 1.f;
    } else if (f == 466) {
      val = dihedral_f(sXi[2], sXnb[e][0], sXnb[e][1], sXnb[e][2]);  // ir_phi
    } else if (f == 467) {
      val = dihedral_f(sXi[0], sXi[1], sXi[2], sXnb[e][0]);          // ir_psi
    }
    sA[e * ARS + f] = f2bf(val);
  }
  __syncthreads();

  // 16 waves: wave w owns output tile (mi = w/8, nj = w%8). 15 K-steps of
  // v_wmma_f32_16x16x32_bf16. A from LDS (ds_load_b128 x2), B from L2-resident
  // transposed bf16 W (global_load_b128 x2). ISA 16-bit A layout:
  //   lanes 0-15: K = k0+0..7 then k0+16..23 ; lanes 16-31: +8.
  int lane = tid & 31;
  int w    = tid >> 5;
  int miT  = w >> 3, njT = w & 7;
  int arow = miT * 16 + (lane & 15);
  int ncol = njT * 16 + (lane & 15);
  int dsel = (lane & 16) ? 8 : 0;
  v8f c = {};
  for (int k0 = 0; k0 < EPAD; k0 += 32) {
    ABfrag a, bb;
    const unsigned short* ap = &sA[arow * ARS + k0 + dsel];
    a.f[0] = *(const f32x4*)(ap);
    a.f[1] = *(const f32x4*)(ap + 16);
    const unsigned short* bp = &wT[ncol * EPAD + k0 + dsel];
    bb.f[0] = *(const f32x4*)(bp);
    bb.f[1] = *(const f32x4*)(bp + 16);
    c = __builtin_amdgcn_wmma_f32_16x16x32_bf16(false, a.v, false, bb.v,
                                                (short)0, c, false, false);
  }
  #pragma unroll
  for (int r = 0; r < 8; ++r) {
    int m = miT * 16 + ((lane < 16) ? r : r + 8);   // ISA 16x16 f32 C/D layout
    sC[m * EDGE_F_ + ncol] = c[r];
  }
  __syncthreads();

  // bias + LayerNorm over 128 features; 16 lanes per edge row,
  // reduction stays inside a 16-lane half of the wave32.
  int row = tid >> 4, t16 = tid & 15;
  float s1 = 0.f, s2 = 0.f;
  for (int q = 0; q < 8; ++q) {
    int n = t16 + q * 16;
    float v = sC[row * EDGE_F_ + n] + be[n];
    s1 += v; s2 += v * v;
  }
  for (int md = 8; md > 0; md >>= 1) {
    s1 += __shfl_xor(s1, md, 32);
    s2 += __shfl_xor(s2, md, 32);
  }
  float mean = s1 * (1.f / 128.f);
  float var  = s2 * (1.f / 128.f) - mean * mean;
  float inv  = rsqrtf(var + 1e-5f);
  float* outp = hE + ((size_t)bi * K_ + row) * EDGE_F_;
  for (int q = 0; q < 8; ++q) {
    int n = t16 + q * 16;
    float v = sC[row * EDGE_F_ + n] + be[n];
    outp[n] = (v - mean) * inv * ge[n] + betae[n];
  }
}

extern "C" void kernel_launch(void* const* d_in, const int* in_sizes, int n_in,
                              void* d_out, int out_size, void* d_ws, size_t ws_size,
                              hipStream_t stream) {
  (void)in_sizes; (void)n_in; (void)out_size; (void)ws_size;
  const float* X     = (const float*)d_in[0];
  const int*   S     = (const int*)d_in[1];
  const float* BB    = (const float*)d_in[2];
  const float* SC    = (const float*)d_in[3];
  const int*   chain = (const int*)d_in[4];
  const float* mask  = (const float*)d_in[5];
  const float* Wn    = (const float*)d_in[6];
  const float* bn    = (const float*)d_in[7];
  const float* gn    = (const float*)d_in[8];
  const float* betan = (const float*)d_in[9];
  const float* We    = (const float*)d_in[10];
  const float* beb   = (const float*)d_in[11];
  const float* geb   = (const float*)d_in[12];
  const float* betae = (const float*)d_in[13];

  float* out   = (float*)d_out;
  float* hV    = out;
  float* hE    = hV + (size_t)B_ * L_ * NODE_F_;
  float* eidxf = hE + (size_t)B_ * L_ * K_ * EDGE_F_;
  float* outX  = eidxf + (size_t)B_ * L_ * K_;

  int* ws_eidx = (int*)d_ws;
  unsigned short* wT =
      (unsigned short*)((char*)d_ws + (size_t)B_ * L_ * K_ * sizeof(int));

  wconv_kernel<<<(EDGE_F_ * EPAD + 255) / 256, 256, 0, stream>>>(We, wT);
  topk_kernel<<<B_ * L_, 256, 0, stream>>>(X, mask, ws_eidx, eidxf);
  node_kernel<<<B_ * L_, 128, 0, stream>>>(S, BB, SC, Wn, bn, gn, betan, hV);
  copyx_kernel<<<(B_ * L_ * 12 + 255) / 256, 256, 0, stream>>>(X, outX, B_ * L_ * 12);
  edge_kernel<<<B_ * L_, 512, 0, stream>>>(X, chain, ws_eidx, wT, beb, geb, betae, hE);
}
